// TGR_76192719831637
// MI455X (gfx1250) — matrix operand order repaired and checked by reading, not compiled, not run on hardware
//
#include <hip/hip_runtime.h>
#include <cstdint>
#include <cstddef>

// ---------------------------------------------------------------------------
// GATv2 x3 + pool + classifier for MI455X (gfx1250, wave32, WMMA)
// ---------------------------------------------------------------------------

typedef __bf16 bf16_t;
typedef bf16_t v16bf __attribute__((ext_vector_type(16)));
typedef bf16_t v8bf  __attribute__((ext_vector_type(8)));
typedef float  v8f   __attribute__((ext_vector_type(8)));

namespace {
constexpr int kN    = 50000;
constexpr int kE    = 600000;
constexpr int kFin  = 128;
constexpr int kH    = 4;
constexpr int kC    = 64;
constexpr int kHC   = 256;
constexpr int kG    = 64;
constexpr int kCls  = 3;
constexpr int kEtot = kE + kN;           // edges + self loops
constexpr int kMT   = kN / 16;           // 3125 row tiles
}

// ---- float atomic max via sign-split monotonic int trick -------------------
__device__ __forceinline__ void atomicMaxF(float* addr, float v) {
  if (v >= 0.0f) atomicMax(reinterpret_cast<int*>(addr), __float_as_int(v));
  else           atomicMin(reinterpret_cast<unsigned int*>(addr), __float_as_uint(v));
}

// ---- small utility kernels -------------------------------------------------
__global__ void k_fill_f32(float* __restrict__ p, float v, int n) {
  int i = blockIdx.x * blockDim.x + threadIdx.x;
  if (i < n) p[i] = v;
}

// fp32 -> bf16 activation convert, optional fused ReLU (between layers)
__global__ void k_cvt_bf16(const float* __restrict__ in, bf16_t* __restrict__ out,
                           int n, int relu) {
  int i = blockIdx.x * blockDim.x + threadIdx.x;
  if (i >= n) return;
  float v = in[i];
  if (relu) v = fmaxf(v, 0.0f);
  out[i] = (bf16_t)v;
}

// Pack W (fin x 256 row-major fp32) into per-lane WMMA B-fragment order:
// P[kt][nt][lane][elem], elem e of lane l maps to K = kt*32 + (l>=16)*16 + e,
// N = nt*16 + (l&15).  (contiguous k order, matching the A loads in k_gemm_dual)
__global__ void k_pack_w(const float* __restrict__ W, bf16_t* __restrict__ P, int fin) {
  int idx = blockIdx.x * blockDim.x + threadIdx.x;
  int total = fin * kHC;
  if (idx >= total) return;
  int elem =  idx        & 15;
  int lane = (idx >> 4)  & 31;
  int nt   = (idx >> 9)  & 15;
  int kt   =  idx >> 13;
  int nn = nt * 16 + (lane & 15);
  int kk = kt * 32 + ((lane >> 4) << 4) + elem;
  P[idx] = (bf16_t)W[kk * kHC + nn];
}

// Dual GEMM: Xl = A@Wl + bl, Xr = A@Wr + br.  One wave -> one 16x16 tile of
// each output; A fragment (16x32 bf16) shared across both WMMAs.
__global__ void k_gemm_dual(const bf16_t* __restrict__ A,    // [n, fin] bf16 row-major
                            const bf16_t* __restrict__ Blp,  // packed fragments
                            const bf16_t* __restrict__ Brp,
                            const float*  __restrict__ bl,
                            const float*  __restrict__ br,
                            float* __restrict__ Xl,          // [n, 256]
                            float* __restrict__ Xr,
                            int fin) {
  const int lane  = threadIdx.x & 31;
  const int wave  = threadIdx.x >> 5;
  const int mTile = blockIdx.x * (blockDim.x >> 5) + wave;
  if (mTile >= kMT) return;                 // wave-uniform: EXEC stays all-ones
  const int nTile  = blockIdx.y;            // 0..15
  const int hi     = lane >> 4;             // lane half selects K sub-block
  const int row    = (mTile << 4) + (lane & 15);
  const int kTiles = fin >> 5;

  const bf16_t* arow = A + (size_t)row * fin;

  v8f accL = {};
  v8f accR = {};
  for (int kt = 0; kt < kTiles; ++kt) {
    // A 16x32 fragment: elems 0..15 -> K = kt*32 + hi*16 .. +15
    const int kA = (kt << 5) + (hi << 4);
    v8bf alo = *(const v8bf*)(arow + kA);
    v8bf ahi = *(const v8bf*)(arow + kA + 8);
    v16bf a  = __builtin_shufflevector(alo, ahi,
                 0,1,2,3,4,5,6,7,8,9,10,11,12,13,14,15);

    const size_t boff = ((((size_t)kt * 16 + nTile) * 32) + lane) * 16;
    v16bf bL = *(const v16bf*)(Blp + boff);
    v16bf bR = *(const v16bf*)(Brp + boff);
    if (kt + 1 < kTiles) {  // pull next B tile toward L2/L0
      const size_t nboff = (((((size_t)kt + 1) * 16 + nTile) * 32) + lane) * 16;
      __builtin_prefetch(Blp + nboff, 0, 1);
      __builtin_prefetch(Brp + nboff, 0, 1);
    }

    accL = __builtin_amdgcn_wmma_f32_16x16x32_bf16(
               false, a, false, bL, (short)0, accL, false, false);
    accR = __builtin_amdgcn_wmma_f32_16x16x32_bf16(
               false, a, false, bR, (short)0, accR, false, false);
  }

  // C/D layout: VGPR r, lanes<16 -> M=r ; lanes>=16 -> M=8+r ; N=lane&15
  const int col = (nTile << 4) + (lane & 15);
  const float bLs = bl[col];
  const float bRs = br[col];
  const size_t base = ((size_t)(mTile << 4) + (hi << 3)) * kHC + col;
#pragma unroll
  for (int r = 0; r < 8; ++r) {
    Xl[base + (size_t)r * kHC] = accL[r] + bLs;
    Xr[base + (size_t)r * kHC] = accR[r] + bRs;
  }
}

// out[n][t] = bias[t]  (aggregation then atomically accumulates on top)
__global__ void k_init_out_bias(float* __restrict__ out,
                                const float* __restrict__ bias, int n) {
  int i = blockIdx.x * blockDim.x + threadIdx.x;
  if (i < n) out[i] = bias[i & (kHC - 1)];
}

// One wave per (edge, head): score = sum_c att[h][c]*leaky(xl[src]+xr[dst])
__global__ void k_edge_score(const float* __restrict__ xl,
                             const float* __restrict__ xr,
                             const float* __restrict__ att,
                             const long long* __restrict__ ei,
                             float* __restrict__ score,
                             float* __restrict__ mx) {
  const int e = blockIdx.x;
  long long s, d;
  if (e < kE) { s = ei[e]; d = ei[kE + e]; } else { s = d = e - kE; }
  const int h = threadIdx.x >> 5;   // head (blockDim = 128 = 4 waves)
  const int l = threadIdx.x & 31;
  const int c = l << 1;             // two channels per lane

  const float* pl = xl + (size_t)s * kHC + h * kC + c;
  const float* pr = xr + (size_t)d * kHC + h * kC + c;
  float v0 = pl[0] + pr[0];
  float v1 = pl[1] + pr[1];
  v0 = v0 > 0.0f ? v0 : 0.2f * v0;
  v1 = v1 > 0.0f ? v1 : 0.2f * v1;
  float p = v0 * att[h * kC + c] + v1 * att[h * kC + c + 1];
#pragma unroll
  for (int off = 16; off > 0; off >>= 1) p += __shfl_down(p, off, 32);
  if (l == 0) {
    score[(size_t)e * kH + h] = p;
    atomicMaxF(mx + (size_t)d * kH + h, p);
  }
}

// ex = exp(score - mx[dst]); in-place; denom[dst] += ex
__global__ void k_edge_exp(float* __restrict__ score,
                           const float* __restrict__ mx,
                           float* __restrict__ denom,
                           const long long* __restrict__ ei) {
  int i = blockIdx.x * blockDim.x + threadIdx.x;     // over Etot*H
  if (i >= kEtot * kH) return;
  int e = i >> 2, h = i & 3;
  long long d = (e < kE) ? ei[kE + e] : (long long)(e - kE);
  float v = __expf(score[i] - mx[(size_t)d * kH + h]);
  score[i] = v;
  atomicAdd(denom + (size_t)d * kH + h, v);
}

// out[dst] += (ex/denom[dst]) * xl[src] ; one thread = 4 channels (float4)
__global__ void k_edge_aggr(const float* __restrict__ xl,
                            const float* __restrict__ ex,
                            const float* __restrict__ denom,
                            const long long* __restrict__ ei,
                            float* __restrict__ out) {
  long long i = (long long)blockIdx.x * blockDim.x + threadIdx.x;
  if (i >= (long long)kEtot * 64) return;
  int e = (int)(i >> 6), q = (int)(i & 63);
  long long s, d;
  if (e < kE) { s = ei[e]; d = ei[kE + e]; } else { s = d = e - kE; }
  int h = q >> 4;
  float alpha = ex[(size_t)e * kH + h] / denom[(size_t)d * kH + h];
  const float4 v = *(const float4*)(xl + (size_t)s * kHC + q * 4);
  float* o = out + (size_t)d * kHC + q * 4;
  atomicAdd(o + 0, alpha * v.x);
  atomicAdd(o + 1, alpha * v.y);
  atomicAdd(o + 2, alpha * v.z);
  atomicAdd(o + 3, alpha * v.w);
}

// mean+max pooling accumulators over graphs
__global__ void k_pool(const float* __restrict__ h,
                       const long long* __restrict__ batch,
                       float* __restrict__ gsum, float* __restrict__ gmax,
                       float* __restrict__ gcnt) {
  int i = blockIdx.x * blockDim.x + threadIdx.x;     // over N*HC
  if (i >= kN * kHC) return;
  int node = i >> 8, t = i & (kHC - 1);
  long long g = batch[node];
  float v = h[i];
  atomicAdd(gsum + (size_t)g * kHC + t, v);
  atomicMaxF(gmax + (size_t)g * kHC + t, v);
  if (t == 0) atomicAdd(gcnt + g, 1.0f);
}

// p = mean + max ; logits = p @ Wout + bout ; softmax.  One block per graph.
__global__ void k_head(const float* __restrict__ gsum, const float* __restrict__ gmax,
                       const float* __restrict__ gcnt,
                       const float* __restrict__ Wout, const float* __restrict__ bout,
                       float* __restrict__ out) {
  const int g = blockIdx.x;
  const int t = threadIdx.x;                         // 256 threads
  __shared__ float sm[kHC];
  float cnt = fmaxf(gcnt[g], 1.0f);
  float p = gsum[(size_t)g * kHC + t] / cnt + gmax[(size_t)g * kHC + t];
  float lg[kCls];
  for (int c = 0; c < kCls; ++c) {
    sm[t] = p * Wout[t * kCls + c];
    __syncthreads();
    for (int off = kHC / 2; off > 0; off >>= 1) {
      if (t < off) sm[t] += sm[t + off];
      __syncthreads();
    }
    lg[c] = sm[0] + bout[c];
    __syncthreads();
  }
  if (t == 0) {
    float m = fmaxf(lg[0], fmaxf(lg[1], lg[2]));
    float e0 = __expf(lg[0] - m), e1 = __expf(lg[1] - m), e2 = __expf(lg[2] - m);
    float s = e0 + e1 + e2;
    out[g * kCls + 0] = e0 / s;
    out[g * kCls + 1] = e1 / s;
    out[g * kCls + 2] = e2 / s;
  }
}

// ---------------------------------------------------------------------------
static inline size_t al256(size_t x) { return (x + 255) & ~(size_t)255; }

extern "C" void kernel_launch(void* const* d_in, const int* in_sizes, int n_in,
                              void* d_out, int out_size, void* d_ws, size_t ws_size,
                              hipStream_t stream) {
  (void)in_sizes; (void)n_in; (void)out_size; (void)ws_size;
  const float*     x     = (const float*)d_in[0];
  const long long* ei    = (const long long*)d_in[1];
  const long long* batch = (const long long*)d_in[2];
  const float *Wl[3], *bl[3], *Wr[3], *br[3], *att[3], *bias[3];
  for (int li = 0; li < 3; ++li) {
    Wl[li]   = (const float*)d_in[3 + 6 * li + 0];
    bl[li]   = (const float*)d_in[3 + 6 * li + 1];
    Wr[li]   = (const float*)d_in[3 + 6 * li + 2];
    br[li]   = (const float*)d_in[3 + 6 * li + 3];
    att[li]  = (const float*)d_in[3 + 6 * li + 4];
    bias[li] = (const float*)d_in[3 + 6 * li + 5];
  }
  const float* Wout = (const float*)d_in[21];
  const float* bout = (const float*)d_in[22];
  float* out = (float*)d_out;

  // ---- workspace carve-up (~141 MB) ---------------------------------------
  char* w = (char*)d_ws;
  size_t off = 0;
  float*  xlbuf = (float*)(w + off);  off += al256((size_t)kN * kHC * 4);     // 51.2 MB
  float*  hbuf  = (float*)(w + off);  off += al256((size_t)kN * kHC * 4);     // xr/out/h
  bf16_t* Abf   = (bf16_t*)(w + off); off += al256((size_t)kN * kHC * 2);     // 25.6 MB
  bf16_t* Blp   = (bf16_t*)(w + off); off += al256((size_t)kHC * kHC * 2);
  bf16_t* Brp   = (bf16_t*)(w + off); off += al256((size_t)kHC * kHC * 2);
  float*  scor  = (float*)(w + off);  off += al256((size_t)kEtot * kH * 4);   // 10.4 MB
  float*  mx    = (float*)(w + off);  off += al256((size_t)kN * kH * 4);
  float*  den   = (float*)(w + off);  off += al256((size_t)kN * kH * 4);
  float*  gsum  = (float*)(w + off);  off += al256((size_t)kG * kHC * 4);
  float*  gmax  = (float*)(w + off);  off += al256((size_t)kG * kHC * 4);
  float*  gcnt  = (float*)(w + off);  off += al256((size_t)kG * 4);

  const int TPB = 256;
  auto nb = [](long long n, int tpb) { return (unsigned)((n + tpb - 1) / tpb); };

  for (int li = 0; li < 3; ++li) {
    const int fin = (li == 0) ? kFin : kHC;
    const int tot = kN * fin;

    // activations -> bf16 (ReLU fused for layers 1,2); layer0 reads x
    k_cvt_bf16<<<nb(tot, TPB), TPB, 0, stream>>>(li == 0 ? x : hbuf, Abf, tot, li > 0);
    // weight fragment packing (deterministic, cheap: fin*256 elements each)
    k_pack_w<<<nb(fin * kHC, TPB), TPB, 0, stream>>>(Wl[li], Blp, fin);
    k_pack_w<<<nb(fin * kHC, TPB), TPB, 0, stream>>>(Wr[li], Brp, fin);

    // fused dual GEMM on WMMA: xl -> xlbuf, xr -> hbuf
    dim3 ggrid(nb(kMT, 8), kHC / 16);
    k_gemm_dual<<<ggrid, 256, 0, stream>>>(Abf, Blp, Brp, bl[li], br[li],
                                           xlbuf, hbuf, fin);

    // softmax state
    k_fill_f32<<<nb(kN * kH, TPB), TPB, 0, stream>>>(mx, -INFINITY, kN * kH);
    k_fill_f32<<<nb(kN * kH, TPB), TPB, 0, stream>>>(den, 0.0f, kN * kH);

    // per-edge attention
    k_edge_score<<<kEtot, 128, 0, stream>>>(xlbuf, hbuf, att[li], ei, scor, mx);
    k_edge_exp<<<nb((long long)kEtot * kH, TPB), TPB, 0, stream>>>(scor, mx, den, ei);

    // out aliases xr (hbuf) — xr is dead after k_edge_score
    k_init_out_bias<<<nb((long long)kN * kHC, TPB), TPB, 0, stream>>>(hbuf, bias[li], kN * kHC);
    k_edge_aggr<<<nb((long long)kEtot * 64, TPB), TPB, 0, stream>>>(xlbuf, scor, den, ei, hbuf);
    // next layer's h is hbuf
  }

  // pooling + classifier
  k_fill_f32<<<nb(kG * kHC, TPB), TPB, 0, stream>>>(gsum, 0.0f, kG * kHC);
  k_fill_f32<<<nb(kG * kHC, TPB), TPB, 0, stream>>>(gmax, -INFINITY, kG * kHC);
  k_fill_f32<<<1, kG, 0, stream>>>(gcnt, 0.0f, kG);
  k_pool<<<nb((long long)kN * kHC, TPB), TPB, 0, stream>>>(hbuf, batch, gsum, gmax, gcnt);
  k_head<<<kG, kHC, 0, stream>>>(gsum, gmax, gcnt, Wout, bout, out);
}
